// MultiheadAttention_14645838480103
// MI455X (gfx1250) — matrix operand order repaired and checked by reading, not compiled, not run on hardware
//
#include <hip/hip_runtime.h>

// ---------------------------------------------------------------------------
// Complex multi-head attention for MI455X (gfx1250, wave32, WMMA).
// All matmuls: bf16 WMMA (v_wmma_f32_16x16x32_bf16) with f32 accumulation.
// Tile staging: GLOBAL_LOAD_ASYNC_TO_LDS_B128 (ASYNCcnt) when available, so
// no VGPRs are burned on global->LDS staging and nothing spills to scratch.
// attn_weights (256MB, > 192MB L2) are moved with non-temporal accesses.
// ---------------------------------------------------------------------------

typedef unsigned short u16;
typedef __attribute__((ext_vector_type(16))) __bf16 v16bf;
typedef __attribute__((ext_vector_type(8)))  float  v8f;
typedef int v4i __attribute__((vector_size(16)));   // matches builtin pointee type

#define LDS_PITCH 40  // ushorts per LDS row: 32 bf16 data + 8 pad (80B) -> conflict-free b128 reads

#if defined(__AMDGCN__) && __has_builtin(__builtin_amdgcn_global_load_async_to_lds_b128)
#define USE_ASYNC 1
#else
#define USE_ASYNC 0
#endif

#if USE_ASYNC
__device__ __forceinline__ void async_copy16(const u16* g, u16* l) {
  // Builtin wants (v4i AS1*, v4i AS3*, imm offset, imm cpol).
  // Flat LDS pointers carry the LDS byte offset in the low 32 bits (ISA 10.2).
  auto gp = (v4i __attribute__((address_space(1)))*)(unsigned long long)(const void*)g;
  auto lp = (v4i __attribute__((address_space(3)))*)(unsigned)(unsigned long long)(void*)l;
  __builtin_amdgcn_global_load_async_to_lds_b128(gp, lp, 0, 0);
}

__device__ __forceinline__ void wait_async0() {
#if __has_builtin(__builtin_amdgcn_s_wait_asynccnt)
  __builtin_amdgcn_s_wait_asynccnt(0);
#else
  asm volatile("s_wait_asynccnt 0x0" ::: "memory");
#endif
}
#endif

__device__ __forceinline__ u16 f32_bf16(float f) {
  unsigned u = __float_as_uint(f);
  return (u16)((u + 0x7FFFu + ((u >> 16) & 1u)) >> 16);  // RNE
}

union FragU { uint4 u[2]; v16bf v; };

// A-matrix 16x32 bf16 fragment (ISA 7.12.2): lane = row; half-wave selects K
// groups {0..7,16..23} / {8..15,24..31}.
__device__ __forceinline__ v16bf lds_frag_a(const u16* plane, int row, int lane) {
  const int half = lane >> 4, idx = lane & 15;
  const u16* b = plane + (row + idx) * LDS_PITCH + half * 8;
  FragU f;
  f.u[0] = *(const uint4*)(b);
  f.u[1] = *(const uint4*)(b + 16);
  return f.v;
}

// B-matrix 32x16 bf16 fragment: lane = column; half-wave selects K
// groups {0..7,16..23} / {8..15,24..31} (same order as A).
__device__ __forceinline__ v16bf lds_frag_b(const u16* plane, int row, int lane) {
  const int half = lane >> 4, idx = lane & 15;
  const u16* b = plane + (row + idx) * LDS_PITCH + half * 8;
  FragU f;
  f.u[0] = *(const uint4*)(b);
  f.u[1] = *(const uint4*)(b + 16);
  return f.v;
}

__device__ __forceinline__ v16bf frag_neg(v16bf x) {
  FragU f; f.v = x;
#pragma unroll
  for (int i = 0; i < 2; ++i) {
    f.u[i].x ^= 0x80008000u; f.u[i].y ^= 0x80008000u;
    f.u[i].z ^= 0x80008000u; f.u[i].w ^= 0x80008000u;
  }
  return f.v;
}

struct GemmP {
  const u16* Ar; const u16* Ai;      // A planes, row-major [M,K], leading dim ldA
  const u16* Br; const u16* Bi;      // B planes, row-major [N,K], leading dim ldB (NT gemm)
  long ldA, ldB, ldC;
  long aOut, aIn, bOut, bIn, cOut, cIn;  // batch offsets: off = (z>>4)*Out + (z&15)*In
  int  K;
  int  hasBias;
  int  storePx;                       // 1: interleaved complex f32 (NT stores); 0: bf16 planes
  float alpha;
  const float* biasR; const float* biasI;
  float2* Cpx;
  u16* CbR; u16* CbI;
};

// Block = 256 threads = 8 waves (WM x WN wave grid). Each wave: 32x32 complex
// tile = 2x2 WMMA tiles x (re,im) = 8 v8f accumulators.
// CONJ: C = A . conj(B)^T (sign folded into the Bi fragment at load).
template <int WM, int WN, bool CONJ>
__global__ __launch_bounds__(256) void cgemm_nt(GemmP p) {
  constexpr int TM = WM * 32, TN = WN * 32;
  constexpr int AITER = TM * 4 / 256;   // 16B chunks per thread per A plane
  constexpr int BITER = TN * 4 / 256;
  constexpr int oAr = 0;
  constexpr int oAi = TM * LDS_PITCH;
  constexpr int oBr = 2 * TM * LDS_PITCH;
  constexpr int oBi = (2 * TM + TN) * LDS_PITCH;

  __shared__ u16 smem[2][(2 * TM + 2 * TN) * LDS_PITCH];

  const int tid  = threadIdx.x;
  const int lane = tid & 31;
  const int wave = tid >> 5;
  const int wm   = wave % WM;
  const int wn   = wave / WM;
  const int mBase = blockIdx.y * TM;
  const int nBase = blockIdx.x * TN;

  const long zo = (long)(blockIdx.z >> 4), zi = (long)(blockIdx.z & 15);
  const u16* gAr = p.Ar + zo * p.aOut + zi * p.aIn;
  const u16* gAi = p.Ai + zo * p.aOut + zi * p.aIn;
  const u16* gBr = p.Br + zo * p.bOut + zi * p.bIn;
  const u16* gBi = p.Bi + zo * p.bOut + zi * p.bIn;
  const long cOff = zo * p.cOut + zi * p.cIn;

  v8f accR[2][2], accI[2][2];
  const v8f vzero = {0.f, 0.f, 0.f, 0.f, 0.f, 0.f, 0.f, 0.f};
#pragma unroll
  for (int i = 0; i < 2; ++i)
#pragma unroll
    for (int j = 0; j < 2; ++j) { accR[i][j] = vzero; accI[i][j] = vzero; }

  // Stage one 32-deep K slab of all four planes into LDS buffer `buf`.
  auto stage_to_lds = [&](int ks, int buf) {
    const long kB = (long)ks * 32;
    u16* s = smem[buf];
#pragma unroll
    for (int i = 0; i < AITER; ++i) {
      const int c = tid + i * 256;
      const long off = (long)(mBase + (c >> 2)) * p.ldA + kB + (long)(c & 3) * 8;
      const int lo = (c >> 2) * LDS_PITCH + (c & 3) * 8;
#if USE_ASYNC
      async_copy16(gAr + off, s + oAr + lo);
      async_copy16(gAi + off, s + oAi + lo);
#else
      *(uint4*)(s + oAr + lo) = *(const uint4*)(gAr + off);
      *(uint4*)(s + oAi + lo) = *(const uint4*)(gAi + off);
#endif
    }
#pragma unroll
    for (int i = 0; i < BITER; ++i) {
      const int c = tid + i * 256;
      const long off = (long)(nBase + (c >> 2)) * p.ldB + kB + (long)(c & 3) * 8;
      const int lo = (c >> 2) * LDS_PITCH + (c & 3) * 8;
#if USE_ASYNC
      async_copy16(gBr + off, s + oBr + lo);
      async_copy16(gBi + off, s + oBi + lo);
#else
      *(uint4*)(s + oBr + lo) = *(const uint4*)(gBr + off);
      *(uint4*)(s + oBi + lo) = *(const uint4*)(gBi + off);
#endif
    }
  };

#if !USE_ASYNC
  auto prefetch_stage = [&](int ks) {
    const long kB = (long)ks * 32;
#pragma unroll
    for (int i = 0; i < AITER; ++i) {
      const int c = tid + i * 256;
      const long off = (long)(mBase + (c >> 2)) * p.ldA + kB + (long)(c & 3) * 8;
      __builtin_prefetch(gAr + off, 0, 1);
      __builtin_prefetch(gAi + off, 0, 1);
    }
#pragma unroll
    for (int i = 0; i < BITER; ++i) {
      const int c = tid + i * 256;
      const long off = (long)(nBase + (c >> 2)) * p.ldB + kB + (long)(c & 3) * 8;
      __builtin_prefetch(gBr + off, 0, 1);
      __builtin_prefetch(gBi + off, 0, 1);
    }
  };
#endif

  // Cr = Ar*Br + (-Ai)*Bi'  ;  Ci = Ai*Br + Ar*Bi'   (Bi' = CONJ ? -Bi : Bi)
  auto compute = [&](int buf) {
    const u16* s = smem[buf];
    v16bf fBr[2], fBi[2];
#pragma unroll
    for (int nt = 0; nt < 2; ++nt) {
      const int row = wn * 32 + nt * 16;
      fBr[nt] = lds_frag_b(s + oBr, row, lane);
      v16bf bi = lds_frag_b(s + oBi, row, lane);
      fBi[nt] = CONJ ? frag_neg(bi) : bi;
    }
#pragma unroll
    for (int mt = 0; mt < 2; ++mt) {
      const int row = wm * 32 + mt * 16;
      const v16bf fAr  = lds_frag_a(s + oAr, row, lane);
      const v16bf fAi  = lds_frag_a(s + oAi, row, lane);
      const v16bf fAiN = frag_neg(fAi);
#pragma unroll
      for (int nt = 0; nt < 2; ++nt) {
        accR[mt][nt] = __builtin_amdgcn_wmma_f32_16x16x32_bf16(
            false, fAr, false, fBr[nt], (short)0, accR[mt][nt], false, false);
        accR[mt][nt] = __builtin_amdgcn_wmma_f32_16x16x32_bf16(
            false, fAiN, false, fBi[nt], (short)0, accR[mt][nt], false, false);
        accI[mt][nt] = __builtin_amdgcn_wmma_f32_16x16x32_bf16(
            false, fAi, false, fBr[nt], (short)0, accI[mt][nt], false, false);
        accI[mt][nt] = __builtin_amdgcn_wmma_f32_16x16x32_bf16(
            false, fAr, false, fBi[nt], (short)0, accI[mt][nt], false, false);
      }
    }
  };

  const int nK = p.K / 32;
  stage_to_lds(0, 0);

#if USE_ASYNC
  for (int ks = 0; ks < nK; ++ks) {
    wait_async0();            // our async LDS writes landed
    __syncthreads();          // everyone's landed; prev readers done
    if (ks + 1 < nK) stage_to_lds(ks + 1, (ks + 1) & 1);  // DMA overlaps WMMAs
    compute(ks & 1);
  }
#else
  for (int ks = 0; ks < nK; ++ks) {
    __syncthreads();
    if (ks + 1 < nK) prefetch_stage(ks + 1);   // global_prefetch, no VGPR cost
    compute(ks & 1);
    if (ks + 1 < nK) stage_to_lds(ks + 1, (ks + 1) & 1);  // short-lived regs
  }
#endif

  // Epilogue. C 16x16 f32 layout: lanes 0-15 -> M=r, lanes 16-31 -> M=8+r, N=lane&15.
  const int rowAdd = (lane >> 4) * 8;
  const int colIdx = lane & 15;
#pragma unroll
  for (int mt = 0; mt < 2; ++mt)
#pragma unroll
    for (int nt = 0; nt < 2; ++nt) {
      const int tRow = mBase + wm * 32 + mt * 16 + rowAdd;
      const int tCol = nBase + wn * 32 + nt * 16 + colIdx;
      float br = 0.f, bi = 0.f;
      if (p.hasBias) { br = p.biasR[tCol]; bi = p.biasI[tCol]; }
#pragma unroll
      for (int r = 0; r < 8; ++r) {
        const long idx = cOff + (long)(tRow + r) * p.ldC + tCol;
        const float cr = accR[mt][nt][r] * p.alpha + br;
        const float ci = accI[mt][nt][r] * p.alpha + bi;
        if (p.storePx) {
          union { float2 f; double d; } u; u.f = make_float2(cr, ci);
          __builtin_nontemporal_store(u.d, (double*)(p.Cpx + idx));
        } else {
          p.CbR[idx] = f32_bf16(cr);
          p.CbI[idx] = f32_bf16(ci);
        }
      }
    }
}

// f32 -> bf16, 4 elems/thread, n multiple of 1024
__global__ __launch_bounds__(256) void cvt_bf16(const float* __restrict__ s,
                                                u16* __restrict__ d, int n) {
  const int i = blockIdx.x * 1024 + threadIdx.x * 4;
  if (i + 3 < n) {
    const float4 v = *(const float4*)(s + i);
    d[i + 0] = f32_bf16(v.x); d[i + 1] = f32_bf16(v.y);
    d[i + 2] = f32_bf16(v.z); d[i + 3] = f32_bf16(v.w);
  }
}

// v[b][s][h*64+d] (bf16 planes) -> vT[bh][d][s] so attn.v is also NT-form.
__global__ __launch_bounds__(256) void vtrans(const u16* __restrict__ vr,
                                              const u16* __restrict__ vi,
                                              u16* __restrict__ tr,
                                              u16* __restrict__ ti) {
  constexpr int S = 1024, D = 1024;
  __shared__ u16 t[64][72];
  const int tid = threadIdx.x;
  const int bh = blockIdx.y, b = bh >> 4, h = bh & 15;
  const int s0 = blockIdx.x * 64;
  for (int pl = 0; pl < 2; ++pl) {
    const u16* src = pl ? vi : vr;
    u16* dst = pl ? ti : tr;
    const int dIn = tid & 63, sIn = tid >> 6;       // coalesced read along d
#pragma unroll 4
    for (int pp = 0; pp < 16; ++pp) {
      const int sl = pp * 4 + sIn;
      t[sl][dIn] = src[(long)b * S * D + (long)(s0 + sl) * D + h * 64 + dIn];
    }
    __syncthreads();
    const int sOut = tid & 63, dOut0 = tid >> 6;    // coalesced write along s
#pragma unroll 4
    for (int pp = 0; pp < 16; ++pp) {
      const int dd = pp * 4 + dOut0;
      dst[(long)bh * 64 * S + (long)dd * S + s0 + sOut] = t[sOut][dd];
    }
    __syncthreads();
  }
}

// attn_weights = softmax(|z|) * z/|z|, in place over interleaved complex rows
// of length 1024 in d_out (non-temporal: 256MB stream > 192MB L2); also emit
// bf16 planes for the attn.v GEMM.
__global__ __launch_bounds__(256) void softmax_sign(float2* __restrict__ w,
                                                    u16* __restrict__ wr,
                                                    u16* __restrict__ wi) {
  __shared__ float red[256];
  const long base = (long)blockIdx.x * 1024;
  const int tid = threadIdx.x;
  float2 zv[4]; float av[4];
  float m = -1e30f;
#pragma unroll
  for (int k = 0; k < 4; ++k) {
    const int j = tid + k * 256;
    union { double d; float2 f; } u;
    u.d = __builtin_nontemporal_load((const double*)(w + base + j));
    zv[k] = u.f;
    av[k] = sqrtf(u.f.x * u.f.x + u.f.y * u.f.y);
    m = fmaxf(m, av[k]);
  }
  red[tid] = m; __syncthreads();
  for (int st = 128; st > 0; st >>= 1) {
    if (tid < st) red[tid] = fmaxf(red[tid], red[tid + st]);
    __syncthreads();
  }
  m = red[0]; __syncthreads();
  float sum = 0.f;
#pragma unroll
  for (int k = 0; k < 4; ++k) sum += expf(av[k] - m);
  red[tid] = sum; __syncthreads();
  for (int st = 128; st > 0; st >>= 1) {
    if (tid < st) red[tid] += red[tid + st];
    __syncthreads();
  }
  const float inv = 1.0f / red[0];
#pragma unroll
  for (int k = 0; k < 4; ++k) {
    const int j = tid + k * 256;
    const float pr = expf(av[k] - m) * inv;
    const float sc = (av[k] > 0.f) ? pr / av[k] : 0.f;  // softmax * z/|z|
    union { float2 f; double d; } uo;
    uo.f = make_float2(zv[k].x * sc, zv[k].y * sc);
    __builtin_nontemporal_store(uo.d, (double*)(w + base + j));
    wr[base + j] = f32_bf16(uo.f.x);
    wi[base + j] = f32_bf16(uo.f.y);
  }
}

extern "C" void kernel_launch(void* const* d_in, const int* in_sizes, int n_in,
                              void* d_out, int out_size, void* d_ws, size_t ws_size,
                              hipStream_t stream) {
  (void)in_sizes; (void)n_in; (void)out_size; (void)ws_size;
  constexpr long B = 2, S = 1024, D = 1024, HD = 64;
  constexpr long ND = B * S * D;          // 2097152 elements
  constexpr long NW = D * D;              // 1048576 elements
  constexpr long NWGT = 32L * S * S;      // 33554432 attn-weight elements

  char* ws = (char*)d_ws;
  size_t off = 0;
  auto alloc = [&](size_t bytes) -> char* {
    char* p = ws + off;
    off += (bytes + 255) & ~(size_t)255;
    return p;
  };

  u16* xbf[6];  for (int i = 0; i < 6; ++i) xbf[i] = (u16*)alloc(ND * 2);
  u16* wbf[8];  for (int i = 0; i < 8; ++i) wbf[i] = (u16*)alloc(NW * 2);
  u16* qb[2];   for (int i = 0; i < 2; ++i) qb[i]  = (u16*)alloc(ND * 2);
  u16* kb[2];   for (int i = 0; i < 2; ++i) kb[i]  = (u16*)alloc(ND * 2);
  u16* vb[2];   for (int i = 0; i < 2; ++i) vb[i]  = (u16*)alloc(ND * 2);
  u16* vt[2];   for (int i = 0; i < 2; ++i) vt[i]  = (u16*)alloc(ND * 2);
  u16* wpl[2];  for (int i = 0; i < 2; ++i) wpl[i] = (u16*)alloc(NWGT * 2);
  u16* ab[2];   for (int i = 0; i < 2; ++i) ab[i]  = (u16*)alloc(ND * 2);

  // --- f32 -> bf16 converts: 6 activations + 8 weight planes
  for (int i = 0; i < 6; ++i)
    cvt_bf16<<<dim3(ND / 1024), dim3(256), 0, stream>>>((const float*)d_in[i], xbf[i], (int)ND);
  const int wIdx[8] = {6, 7, 10, 11, 14, 15, 18, 19};
  for (int i = 0; i < 8; ++i)
    cvt_bf16<<<dim3(NW / 1024), dim3(256), 0, stream>>>((const float*)d_in[wIdx[i]], wbf[i], (int)NW);

  // --- Q/K/V complex linears: [2048,1024] x [1024,1024]^T + bias -> bf16 planes
  for (int l = 0; l < 3; ++l) {
    GemmP p{};
    p.Ar = xbf[2 * l]; p.Ai = xbf[2 * l + 1];
    p.Br = wbf[2 * l]; p.Bi = wbf[2 * l + 1];
    p.ldA = D; p.ldB = D; p.ldC = D;
    p.K = (int)D; p.alpha = 1.0f;
    p.hasBias = 1;
    p.biasR = (const float*)d_in[6 + 4 * l + 2];
    p.biasI = (const float*)d_in[6 + 4 * l + 3];
    p.storePx = 0;
    u16** outp = (l == 0) ? qb : (l == 1) ? kb : vb;
    p.CbR = outp[0]; p.CbI = outp[1];
    cgemm_nt<2, 4, false><<<dim3(D / 128, (B * S) / 64, 1), dim3(256), 0, stream>>>(p);
  }

  // --- scores = (1/8) q . conj(k)^T, 32 heads, written interleaved into d_out
  float2* wOut = (float2*)d_out + ND;  // attn_weights region
  {
    GemmP p{};
    p.Ar = qb[0]; p.Ai = qb[1]; p.Br = kb[0]; p.Bi = kb[1];
    p.ldA = D; p.ldB = D; p.ldC = S;
    p.aOut = S * D; p.aIn = HD;   // off = b*S*D + h*64 (head view, no copy)
    p.bOut = S * D; p.bIn = HD;
    p.cOut = 16L * S * S; p.cIn = S * S;
    p.K = (int)HD; p.alpha = 0.125f;  // HD^-0.5
    p.storePx = 1; p.Cpx = wOut;
    cgemm_nt<2, 4, true><<<dim3(S / 128, S / 64, 32), dim3(256), 0, stream>>>(p);
  }

  // --- softmax(|z|) * sign(z), in place + bf16 planes
  softmax_sign<<<dim3(32 * S), dim3(256), 0, stream>>>(wOut, wpl[0], wpl[1]);

  // --- transpose v per head so attn.v is NT-form
  vtrans<<<dim3(S / 64, 32), dim3(256), 0, stream>>>(vb[0], vb[1], vt[0], vt[1]);

  // --- attn = w @ v : [1024,1024] x [64,1024]^T per head, merge heads on store
  {
    GemmP p{};
    p.Ar = wpl[0]; p.Ai = wpl[1]; p.Br = vt[0]; p.Bi = vt[1];
    p.ldA = S; p.ldB = S; p.ldC = D;
    p.aOut = 16L * S * S; p.aIn = S * S;
    p.bOut = 16L * HD * S; p.bIn = HD * S;
    p.cOut = S * D; p.cIn = HD;   // out[b][t][h*64+d]
    p.K = (int)S; p.alpha = 1.0f;
    p.storePx = 0; p.CbR = ab[0]; p.CbI = ab[1];
    cgemm_nt<4, 2, false><<<dim3(HD / 64, S / 128, 32), dim3(256), 0, stream>>>(p);
  }

  // --- output complex linear -> interleaved complex f32 into d_out[0:ND]
  {
    GemmP p{};
    p.Ar = ab[0]; p.Ai = ab[1]; p.Br = wbf[6]; p.Bi = wbf[7];
    p.ldA = D; p.ldB = D; p.ldC = D;
    p.K = (int)D; p.alpha = 1.0f;
    p.hasBias = 1;
    p.biasR = (const float*)d_in[20]; p.biasI = (const float*)d_in[21];
    p.storePx = 1; p.Cpx = (float2*)d_out;
    cgemm_nt<2, 4, false><<<dim3(D / 128, (B * S) / 64, 1), dim3(256), 0, stream>>>(p);
  }
}
